// SimpleRetention_71262097375678
// MI455X (gfx1250) — matrix operand order repaired and checked by reading, not compiled, not run on hardware
//
#include <hip/hip_runtime.h>
#include <hip/hip_bf16.h>

// ---------------- problem constants (match reference) ----------------
#define Bb   8
#define Cc   2048
#define Ee   256
#define Hh   64
#define NAg  4
#define Tts  (Cc / NAg)        // 512 timesteps
__device__ constexpr float L2K = -0.152003093445266f;  // log2(0.9)

// ---------------- WMMA fragment types (gfx1250, wave32) ----------------
typedef __bf16 bf16;
typedef __attribute__((ext_vector_type(16))) __bf16 v16bf;
typedef __attribute__((ext_vector_type(8)))  __bf16 v8bf;
typedef __attribute__((ext_vector_type(8)))  float  v8f;
typedef int vsi4 __attribute__((vector_size(16)));  // matches builtin proto

static __device__ __forceinline__ v16bf combine8(v8bf lo, v8bf hi) {
    v16bf r;
#pragma unroll
    for (int i = 0; i < 8; ++i) { r[i] = lo[i]; r[i + 8] = hi[i]; }
    return r;
}

static __device__ __forceinline__ v8f wmma_bf16(v16bf a, v16bf b, v8f c) {
    // D(16x16,f32) = A(16x32,bf16) x B(32x16,bf16) + C
    return __builtin_amdgcn_wmma_f32_16x16x32_bf16(
        /*neg_a=*/false, a, /*neg_b=*/false, b,
        /*c_mod=*/(short)0, c, /*reuse_a=*/false, /*reuse_b=*/false);
}

// ---------------- CDNA5 async global->LDS (ASYNCcnt) ----------------
// Copies 16 bytes per lane from global to LDS asynchronously.
static __device__ __forceinline__ void async_b128(const bf16* g, bf16* l) {
#if __has_builtin(__builtin_amdgcn_global_load_async_to_lds_b128)
    __builtin_amdgcn_global_load_async_to_lds_b128(
        (__attribute__((address_space(1))) vsi4*)(size_t)g,
        (__attribute__((address_space(3))) vsi4*)(unsigned)(size_t)l,
        /*offset=*/0, /*cpol=*/0);
#else
    const unsigned           laddr = (unsigned)(size_t)l;  // LDS byte offset
    const unsigned long long gaddr = (unsigned long long)(size_t)g;
    asm volatile("global_load_async_to_lds_b128 %0, %1, off"
                 :: "v"(laddr), "v"(gaddr)
                 : "memory");
#endif
}

template <int N>
static __device__ __forceinline__ void wait_async() {
#if __has_builtin(__builtin_amdgcn_s_wait_asynccnt)
    __builtin_amdgcn_s_wait_asynccnt((short)N);
#else
    asm volatile("s_wait_asynccnt %0" :: "i"(N) : "memory");
#endif
}

// =====================================================================
// Kernel 0: per-batch decay bookkeeping + hstate -> bf16 (transposed)
// =====================================================================
__global__ void prep_kernel(const unsigned char* __restrict__ dones,
                            const float* __restrict__ hstate,
                            int* __restrict__ csum, float* __restrict__ xi,
                            float* __restrict__ dml, float* __restrict__ delta,
                            bf16* __restrict__ hbT) {
    const int b   = blockIdx.x;
    const int tid = threadIdx.x;
    __shared__ int scs[Tts];
    __shared__ int sfirst;
    if (tid == 0) {
        int run = 0, fst = Tts;
        for (int t = 0; t < Tts; ++t) {
            int d = dones[(size_t)b * Cc + t * NAg] ? 1 : 0;
            if (d && fst == Tts) fst = t;
            run += d;
            scs[t] = run;
        }
        sfirst = fst;
        delta[b] = (run == 0) ? 1.0f : 0.0f;
    }
    __syncthreads();
    const int total = scs[Tts - 1];
    for (int t = tid; t < Tts; t += blockDim.x) csum[b * Tts + t] = scs[t];
    for (int c = tid; c < Cc; c += blockDim.x) {
        const int t = c >> 2;
        xi[(size_t)b * Cc + c] =
            (t < sfirst) ? __builtin_exp2f((float)(t + 1) * L2K) : 0.0f;
        dml[(size_t)b * Cc + c] =
            (total - scs[t] == 0) ? __builtin_exp2f((float)(Tts - 1 - t) * L2K) : 0.0f;
    }
    for (int i = tid; i < Hh * Hh; i += blockDim.x) {
        const int e = i >> 6, d = i & 63;  // hstate[b][e][d]
        hbT[(size_t)b * Hh * Hh + d * Hh + e] =
            (bf16)hstate[(size_t)b * Hh * Hh + i];
    }
}

// =====================================================================
// Kernel 1: projection GEMM  [B*C,E] x [E,H] -> bf16
//   TRANS_OUT=1 stores [B,H,C] (used for V).
// =====================================================================
template <int TRANS_OUT>
__global__ void proj_kernel(const float* __restrict__ x,
                            const float* __restrict__ w,
                            bf16* __restrict__ ob) {
    __shared__ bf16 sw[Hh * Ee];  // 32 KB: wT[h*E + e]
    const int tid = threadIdx.x;
    for (int i = tid; i < Ee * Hh; i += 256) {
        const int e = i >> 6, h = i & 63;
        sw[h * Ee + e] = (bf16)w[i];
    }
    __syncthreads();

    const int wid = tid >> 5, lane = tid & 31;
    const int n = lane & 15, hi = lane >> 4;
    const int rbg = blockIdx.x * 8 + wid;  // 16-row block over B*C rows
    const float* xrow = x + ((size_t)rbg * 16 + n) * Ee;  // A row = lane%16

    v8f acc[4] = {};
    for (int c = 0; c < Ee / 32; ++c) {  // K chunks of 32
        v16bf a;
#pragma unroll
        for (int i = 0; i < 8; ++i) {
            a[i]     = (bf16)xrow[c * 32 + hi * 16 + i];
            a[i + 8] = (bf16)xrow[c * 32 + hi * 16 + 8 + i];
        }
#pragma unroll
        for (int j = 0; j < 4; ++j) {
            const bf16* bp = &sw[(j * 16 + n) * Ee + c * 32 + hi * 16];
            const v16bf bfr =
                combine8(*(const v8bf*)bp, *(const v8bf*)(bp + 8));
            acc[j] = wmma_bf16(a, bfr, acc[j]);
        }
    }
#pragma unroll
    for (int r = 0; r < 8; ++r) {
        const int grow = rbg * 16 + hi * 8 + r;  // global row over B*C
        if (TRANS_OUT) {
            const int b = grow >> 11, cidx = grow & (Cc - 1);
#pragma unroll
            for (int j = 0; j < 4; ++j)
                ob[((size_t)b * Hh + j * 16 + n) * Cc + cidx] = (bf16)acc[j][r];
        } else {
            bf16* orow = ob + (size_t)grow * Hh;
#pragma unroll
            for (int j = 0; j < 4; ++j) orow[j * 16 + n] = (bf16)acc[j][r];
        }
    }
}

// =====================================================================
// Kernel 2: retention core.  One wave per 16-row block of one batch.
//   out1[b,rows,:] = sum_{m<=n} (q.k^T * dm) @ v  +  xi * (q @ hstate)
//   k (32x64 bf16) and vt (64x32 bf16) superblock tiles are streamed
//   into per-wave double-buffered LDS with ASYNC global->LDS copies;
//   s_wait_asynccnt gates consumption so the prefetch of block j+1
//   overlaps the WMMA work of block j.
// =====================================================================
__global__ void __launch_bounds__(128)
attn_kernel(const bf16* __restrict__ qb, const bf16* __restrict__ kb,
            const bf16* __restrict__ vt, const bf16* __restrict__ hbT,
            const int* __restrict__ csum, const float* __restrict__ xi,
            float* __restrict__ out1) {
    constexpr int WPB = 4;
    __shared__ __align__(16) bf16 kbuf[WPB][2][32 * Hh];  // 4 KB / buf
    __shared__ __align__(16) bf16 vbuf[WPB][2][Hh * 32];  // 4 KB / buf
    __shared__ __align__(16) bf16 stage[WPB][16 * 32];    // score re-layout

    const int wid = threadIdx.x >> 5, lane = threadIdx.x & 31;
    const int n = lane & 15, hi = lane >> 4;
    const int rbg = blockIdx.x * WPB + wid;  // 0 .. B*128-1
    const int b = rbg >> 7;                  // 128 row-blocks per batch
    const int rb = rbg & 127;

    // per-row constants for the 8 accumulator rows owned by this lane
    int gn[8], csn[8];
    {
        int cs4[4];
#pragma unroll
        for (int i = 0; i < 4; ++i) cs4[i] = csum[b * Tts + rb * 4 + i];
#pragma unroll
        for (int r = 0; r < 8; ++r) {
            const int M = hi * 8 + r;
            gn[r] = rb * 16 + M;
            csn[r] = cs4[M >> 2];
        }
    }

    const bf16* kB = kb + (size_t)b * Cc * Hh;  // [C,H]
    const bf16* vB = vt + (size_t)b * Hh * Cc;  // [H,C]

    // issue async copies of one (k, vt) superblock tile (16 b128 ops)
    auto copy_tiles = [&](int m0, int buf) {
        const bf16* kg = kB + (size_t)m0 * Hh;  // 4 KB contiguous
        bf16* kl = &kbuf[wid][buf][0];
        bf16* vl = &vbuf[wid][buf][0];
#pragma unroll
        for (int i = 0; i < 8; ++i) {
            const int idx = i * 32 + lane;
            async_b128(kg + idx * 8, kl + idx * 8);
            // vt tile: 64 rows x 32 cols, row stride Cc
            async_b128(vB + (size_t)(idx >> 2) * Cc + m0 + (idx & 3) * 8,
                       vl + idx * 8);
        }
    };

    // Q A-fragments (16 x 64 split into two K=32 chunks); A row = lane%16
    v16bf aq[2];
    {
        const bf16* qrow = qb + ((size_t)b * Cc + rb * 16 + n) * Hh;
#pragma unroll
        for (int c = 0; c < 2; ++c) {
            const v8bf lo = *(const v8bf*)(qrow + c * 32 + hi * 16);
            const v8bf hh = *(const v8bf*)(qrow + c * 32 + hi * 16 + 8);
            aq[c] = combine8(lo, hh);
        }
    }

    const int nblk = (rb * 16 + 15) / 32 + 1;
    copy_tiles(0, 0);  // prime the pipeline

    // cross = q @ hstate (overlaps with the first async copy)
    v8f cacc[4] = {};
#pragma unroll
    for (int c = 0; c < 2; ++c)
#pragma unroll
        for (int j = 0; j < 4; ++j) {
            const bf16* hrow =
                hbT + ((size_t)b * Hh + j * 16 + n) * Hh + c * 32 + hi * 16;
            const v16bf bfr =
                combine8(*(const v8bf*)hrow, *(const v8bf*)(hrow + 8));
            cacc[j] = wmma_bf16(aq[c], bfr, cacc[j]);
        }

    // main streaming loop over the lower triangle
    v8f acc[4] = {};
    for (int j32 = 0; j32 < nblk; ++j32) {
        const int m0 = j32 * 32;
        const int cur = j32 & 1;
        if (j32 + 1 < nblk) {
            copy_tiles(m0 + 32, cur ^ 1);  // prefetch next superblock
            wait_async<16>();              // current tile done; prefetch in flight
        } else {
            wait_async<0>();
        }
        const bf16* kt = &kbuf[wid][cur][0];
        const bf16* vti = &vbuf[wid][cur][0];

#pragma unroll
        for (int t = 0; t < 2; ++t) {  // two 16x16 score tiles
            v8f sv = {};
#pragma unroll
            for (int c = 0; c < 2; ++c) {  // B = k^T from LDS, contiguous
                const bf16* krow = kt + (t * 16 + n) * Hh + c * 32 + hi * 16;
                const v16bf bk =
                    combine8(*(const v8bf*)krow, *(const v8bf*)(krow + 8));
                sv = wmma_bf16(aq[c], bk, sv);
            }
            // elementwise decay * episode mask * tril
            const int gm = m0 + t * 16 + n;
            const int tm = gm >> 2;
            const int csm = csum[b * Tts + tm];
#pragma unroll
            for (int r = 0; r < 8; ++r) {
                const int tn = gn[r] >> 2;
                const bool ok = (gn[r] >= gm) && (csn[r] == csm);
                const float d =
                    ok ? __builtin_exp2f((float)(tn - tm) * L2K) : 0.0f;
                sv[r] *= d;
            }
            // stage scores (C-layout -> row-major 16x32 in LDS, bf16)
#pragma unroll
            for (int r = 0; r < 8; ++r)
                stage[wid][(hi * 8 + r) * 32 + t * 16 + n] = (bf16)sv[r];
        }

        // reload staged scores as the A-fragment of S(16x32)
        v16bf sA;
        {
            const bf16* sp = &stage[wid][n * 32];  // A row = lane%16
            sA = combine8(*(const v8bf*)(sp + hi * 16),
                          *(const v8bf*)(sp + hi * 16 + 8));
        }
        // B = V(32x64) from LDS vt tile: K (=c) dim contiguous per lane
#pragma unroll
        for (int j = 0; j < 4; ++j) {
            const bf16* vrow = vti + (j * 16 + n) * 32 + hi * 16;
            const v16bf bv =
                combine8(*(const v8bf*)vrow, *(const v8bf*)(vrow + 8));
            acc[j] = wmma_bf16(sA, bv, acc[j]);
        }
    }

    // epilogue: out = inner + xi * cross
#pragma unroll
    for (int r = 0; r < 8; ++r) {
        const float xr = xi[(size_t)b * Cc + gn[r]];
        float* orow = out1 + ((size_t)b * Cc + gn[r]) * Hh;
#pragma unroll
        for (int j = 0; j < 4; ++j)
            orow[j * 16 + n] = acc[j][r] + xr * cacc[j][r];
    }
}

// =====================================================================
// Kernel 3: next_h[b] = k^T (v * dml) + kappa^T * delta[b] * hstate[b]
// =====================================================================
__global__ void nexth_kernel(const bf16* __restrict__ kb,
                             const bf16* __restrict__ vt,
                             const float* __restrict__ dml,
                             const float* __restrict__ hstate,
                             const float* __restrict__ delta,
                             float* __restrict__ outh) {
    const int b = blockIdx.x;
    __shared__ float sk[32][Hh];
    __shared__ float sv[32][Hh + 1];
    const int tid = threadIdx.x;
    const int h = tid >> 2;          // 0..63
    const int db = (tid & 3) * 16;   // 16-wide d strip
    float acc[16];
#pragma unroll
    for (int i = 0; i < 16; ++i) acc[i] = 0.0f;

    for (int c0 = 0; c0 < Cc; c0 += 32) {
        for (int i = tid; i < 32 * Hh; i += 256) {
            const int cc = i >> 6, col = i & 63;
            sk[cc][col] = (float)kb[((size_t)b * Cc + c0 + cc) * Hh + col];
        }
        for (int i = tid; i < 32 * Hh; i += 256) {
            const int col = i >> 5, cc = i & 31;
            sv[cc][col] = (float)vt[((size_t)b * Hh + col) * Cc + c0 + cc] *
                          dml[(size_t)b * Cc + c0 + cc];
        }
        __syncthreads();
        for (int cc = 0; cc < 32; ++cc) {
            const float kv = sk[cc][h];
#pragma unroll
            for (int dd = 0; dd < 16; ++dd) acc[dd] += kv * sv[cc][db + dd];
        }
        __syncthreads();
    }
    const float chd = __builtin_exp2f((float)Tts * L2K) * delta[b];
#pragma unroll
    for (int dd = 0; dd < 16; ++dd) {
        const size_t o = ((size_t)b * Hh + h) * Hh + db + dd;
        outh[o] = acc[dd] + hstate[o] * chd;
    }
}

// =====================================================================
extern "C" void kernel_launch(void* const* d_in, const int* in_sizes, int n_in,
                              void* d_out, int out_size, void* d_ws,
                              size_t ws_size, hipStream_t stream) {
    const float* key_in = (const float*)d_in[0];
    const float* query  = (const float*)d_in[1];
    const float* value  = (const float*)d_in[2];
    const float* hstate = (const float*)d_in[3];
    const unsigned char* dones = (const unsigned char*)d_in[4];  // bool[B,C]
    const float* w_q = (const float*)d_in[5];
    const float* w_k = (const float*)d_in[6];
    const float* w_v = (const float*)d_in[7];

    float* out1 = (float*)d_out;                       // [B,C,H]
    float* outh = out1 + (size_t)Bb * Cc * Hh;         // [B,H,H]

    // workspace carve-up (~6.3 MB)
    char* ws = (char*)d_ws;
    auto take = [&](size_t bytes) {
        void* p = (void*)ws;
        ws += (bytes + 255) & ~(size_t)255;
        return p;
    };
    bf16* qb  = (bf16*)take((size_t)Bb * Cc * Hh * 2);  // [B,C,H]
    bf16* kb  = (bf16*)take((size_t)Bb * Cc * Hh * 2);  // [B,C,H]
    bf16* vt  = (bf16*)take((size_t)Bb * Hh * Cc * 2);  // [B,H,C] (transposed)
    bf16* hbT = (bf16*)take((size_t)Bb * Hh * Hh * 2);  // [B,H(d),H(e)]
    int*  cs  = (int*) take((size_t)Bb * Tts * 4);
    float* xi  = (float*)take((size_t)Bb * Cc * 4);
    float* dml = (float*)take((size_t)Bb * Cc * 4);
    float* dlt = (float*)take((size_t)Bb * 4);

    prep_kernel<<<Bb, 256, 0, stream>>>(dones, hstate, cs, xi, dml, dlt, hbT);

    proj_kernel<0><<<(Bb * Cc / 16) / 8, 256, 0, stream>>>(query, w_q, qb);
    proj_kernel<0><<<(Bb * Cc / 16) / 8, 256, 0, stream>>>(key_in, w_k, kb);
    proj_kernel<1><<<(Bb * Cc / 16) / 8, 256, 0, stream>>>(value, w_v, vt);

    attn_kernel<<<(Bb * (Cc / 16)) / 4, 128, 0, stream>>>(qb, kb, vt, hbT, cs,
                                                          xi, out1);

    nexth_kernel<<<Bb, 256, 0, stream>>>(kb, vt, dml, hstate, dlt, outh);
}
